// Gemma4Attention_15058155340378
// MI455X (gfx1250) — matrix operand order repaired and checked by reading, not compiled, not run on hardware
//
#include <hip/hip_runtime.h>
#include <hip/hip_bf16.h>

// ---------------------------------------------------------------------------
// Types for CDNA5 WMMA (wave32): V_WMMA_F32_16X16X32_BF16
// ---------------------------------------------------------------------------
typedef __bf16 bf16_t;
typedef __attribute__((ext_vector_type(16))) __bf16 v16bf;
typedef __attribute__((ext_vector_type(8)))  __bf16 v8bf;
typedef __attribute__((ext_vector_type(8)))  float  v8f;

union V16U { v16bf v; struct { v8bf lo; v8bf hi; } h; };

static __device__ __forceinline__ bf16_t f2bf(float f) {
    union { float f; unsigned u; } cv; cv.f = f;
    unsigned r = cv.u + 0x7FFFu + ((cv.u >> 16) & 1u);   // round-to-nearest-even
    unsigned short hs = (unsigned short)(r >> 16);
    return __builtin_bit_cast(bf16_t, hs);
}

static __device__ __forceinline__ v8f wmma_bf16(v16bf a, v16bf b, v8f c) {
    return __builtin_amdgcn_wmma_f32_16x16x32_bf16(
        /*neg_a=*/false, a, /*neg_b=*/false, b,
        /*c_mod=*/(short)0, c, /*reuse_a=*/false, /*reuse_b=*/false);
}

// A-fragment (16x32 bf16) loader (k order chosen here, not the ISA table order): lane m = lane&15, half = lane>>4;
// elems 0..15 <- row[half*16 + 0..15] (contiguous k, matching B fragments)
static __device__ __forceinline__ v16bf load_a_frag(const bf16_t* row, int half) {
    V16U u;
    u.h.lo = *(const v8bf*)(row + half * 16);
    u.h.hi = *(const v8bf*)(row + half * 16 + 8);
    return u.v;
}

#define EMB   2048
#define SEQ   2048
#define BATCH 2
#define NH    16
#define NKV   4
#define HD    128
#define MROWS (BATCH * SEQ)   // 4096

// ---------------------------------------------------------------------------
// One-shot fp32 -> bf16 conversion (8 elements / thread, vector load+store).
// ---------------------------------------------------------------------------
__global__ __launch_bounds__(256)
void f32_to_bf16(const float* __restrict__ in, bf16_t* __restrict__ out, int n8) {
    const int i = blockIdx.x * 256 + threadIdx.x;
    if (i >= n8) return;
    const float4* p = (const float4*)in + (size_t)i * 2;
    const float4 a = p[0], b = p[1];
    v8bf o;
    o[0] = f2bf(a.x); o[1] = f2bf(a.y); o[2] = f2bf(a.z); o[3] = f2bf(a.w);
    o[4] = f2bf(b.x); o[5] = f2bf(b.y); o[6] = f2bf(b.z); o[7] = f2bf(b.w);
    *(v8bf*)(out + (size_t)i * 8) = o;
}

// ---------------------------------------------------------------------------
// Tiled WMMA GEMM: C[M,N] = A[M,K] * B[K,N];  A,B bf16, C f32.  N,K compile
// time so C-store offsets fold into global_store immediates.
// BM=BN=128, BK=32, LDS double-buffered (one barrier per K-slab).
// 256 threads = 8 waves (4x2), each wave computes a 32x64 tile.
// ---------------------------------------------------------------------------
template <int N, int K>
__global__ __launch_bounds__(256)
void gemm_wmma(const bf16_t* __restrict__ Ap, const bf16_t* __restrict__ Bp,
               float* __restrict__ Cp) {
    __shared__ __align__(32) bf16_t As[2][128 * 32];   // [m][k]
    __shared__ __align__(32) bf16_t Bs[2][128 * 32];   // [n][k] (transposed stage)

    const int tid  = threadIdx.x;
    const int m0   = blockIdx.y * 128;
    const int n0   = blockIdx.x * 128;
    const int wave = tid >> 5, lane = tid & 31;
    const int wm = wave >> 1, wn = wave & 1;           // 4 x 2 wave grid
    const int mlo = lane & 15, half = lane >> 4;

    v8f acc[2][4];
#pragma unroll
    for (int i = 0; i < 2; ++i)
#pragma unroll
        for (int j = 0; j < 4; ++j)
            acc[i][j] = (v8f){0.f, 0.f, 0.f, 0.f, 0.f, 0.f, 0.f, 0.f};

    const int arow = tid >> 1, ac0 = (tid & 1) * 16;   // A staging coords
    const int bkk  = tid >> 3, bnn = (tid & 7) * 16;   // B staging coords
    const bf16_t* aBase = Ap + (size_t)(m0 + arow) * K + ac0;
    const bf16_t* bBase = Bp + (size_t)bkk * N + n0 + bnn;

    auto stage = [&](int buf, int k0) {
        // A (128x32): pure 16B vector copies
        const bf16_t* a = aBase + k0;
        *(v8bf*)&As[buf][arow * 32 + ac0]     = *(const v8bf*)a;
        *(v8bf*)&As[buf][arow * 32 + ac0 + 8] = *(const v8bf*)(a + 8);
        __builtin_prefetch((const void*)(a + 64), 0, 1);
        // B (32 k-rows x 128 n): coalesced read, transposed scatter
        const bf16_t* bsrc = bBase + (size_t)k0 * N;
        const v8bf b0 = *(const v8bf*)bsrc;
        const v8bf b1 = *(const v8bf*)(bsrc + 8);
#pragma unroll
        for (int j = 0; j < 8; ++j) Bs[buf][(bnn + j) * 32 + bkk] = b0[j];
#pragma unroll
        for (int j = 0; j < 8; ++j) Bs[buf][(bnn + 8 + j) * 32 + bkk] = b1[j];
        __builtin_prefetch((const void*)(bsrc + (size_t)64 * N), 0, 1);
    };

    auto compute = [&](int buf) {
        v16bf afr[2], bfr[4];
#pragma unroll
        for (int mi = 0; mi < 2; ++mi)
            afr[mi] = load_a_frag(&As[buf][(wm * 32 + mi * 16 + mlo) * 32], half);
#pragma unroll
        for (int ni = 0; ni < 4; ++ni)   // B-frag: lane n=mlo, elems k=half*16+e
            bfr[ni] = *(const v16bf*)&Bs[buf][(wn * 64 + ni * 16 + mlo) * 32 + half * 16];
#pragma unroll
        for (int mi = 0; mi < 2; ++mi)
#pragma unroll
            for (int ni = 0; ni < 4; ++ni)
                acc[mi][ni] = wmma_bf16(afr[mi], bfr[ni], acc[mi][ni]);
    };

    constexpr int NSLAB = K / 32;
    stage(0, 0);
    __syncthreads();
    for (int s = 1; s < NSLAB; ++s) {
        stage(s & 1, s * 32);          // fill next buffer
        compute((s - 1) & 1);          // consume previous buffer
        __syncthreads();               // one barrier per slab
    }
    compute((NSLAB - 1) & 1);

    // ---- epilogue: C-layout lane n = mlo, VGPR v -> row m = v + 8*half ----
    float* cbase = Cp + (size_t)(m0 + wm * 32 + 8 * half) * N + n0 + wn * 64 + mlo;
#pragma unroll
    for (int mi = 0; mi < 2; ++mi)
#pragma unroll
        for (int ni = 0; ni < 4; ++ni)
#pragma unroll
            for (int v = 0; v < 8; ++v)
                cbase[(size_t)(mi * 16 + v) * N + ni * 16] = acc[mi][ni][v];
}

// ---------------------------------------------------------------------------
// RMSNorm (+weight) + RoPE for q/k, RMSNorm for v; fp32 -> bf16 repack.
// q_bf: [b][h][s][d] ; k_bf: [b][kv][s][d] ; v_t: [b][kv][d][s] (transposed!)
// ---------------------------------------------------------------------------
__global__ __launch_bounds__(128)
void norm_rope(const float* __restrict__ q_raw, const float* __restrict__ k_raw,
               const float* __restrict__ v_raw, const float* __restrict__ cosb,
               const float* __restrict__ sinb, const float* __restrict__ qw,
               const float* __restrict__ kw, bf16_t* __restrict__ q_bf,
               bf16_t* __restrict__ k_bf, bf16_t* __restrict__ v_t) {
    const int bs = blockIdx.x;             // b*S + s
    const int s  = bs & (SEQ - 1);
    const int b  = bs >> 11;
    const int hh = blockIdx.y;             // 0..15 q heads, 16..19 k, 20..23 v
    const int d  = threadIdx.x;            // 0..127

    float val;
    if (hh < 16)      val = q_raw[(size_t)bs * 2048 + hh * 128 + d];
    else if (hh < 20) val = k_raw[(size_t)bs * 512 + (hh - 16) * 128 + d];
    else              val = v_raw[(size_t)bs * 512 + (hh - 20) * 128 + d];

    float ss = val * val;
#pragma unroll
    for (int m = 16; m >= 1; m >>= 1) ss += __shfl_xor(ss, m);
    __shared__ float red[4];
    if ((threadIdx.x & 31) == 0) red[threadIdx.x >> 5] = ss;
    __syncthreads();
    const float tot = red[0] + red[1] + red[2] + red[3];
    const float inv = rsqrtf(tot * (1.0f / 128.0f) + 1e-6f);

    float xn = val * inv;
    if (hh < 16)      xn *= qw[d];
    else if (hh < 20) xn *= kw[d];

    __shared__ float xsh[128];
    xsh[d] = xn;
    __syncthreads();

    if (hh < 20) {  // RoPE on q and k
        const float c  = cosb[s * 128 + d];
        const float si = sinb[s * 128 + d];
        const float rot = (d < 64) ? -xsh[d + 64] : xsh[d - 64];
        xn = xn * c + rot * si;
    }

    if (hh < 16)
        q_bf[(((size_t)b * NH + hh) * SEQ + s) * HD + d] = f2bf(xn);
    else if (hh < 20)
        k_bf[(((size_t)b * NKV + (hh - 16)) * SEQ + s) * HD + d] = f2bf(xn);
    else
        v_t[(((size_t)b * NKV + (hh - 20)) * HD + d) * SEQ + s] = f2bf(xn);
}

// ---------------------------------------------------------------------------
// Flash-style causal attention (no 1/sqrt(d) scale, per reference).
// Block = 128 thr (4 waves); wave handles one 16-row Q tile; 32-key blocks.
// Q.K^T: 4x wmma (Kdim=128); P.V: 8x wmma with V pre-transposed [d][s].
// ---------------------------------------------------------------------------
__global__ __launch_bounds__(128)
void attn_wmma(const bf16_t* __restrict__ Q, const bf16_t* __restrict__ Kb,
               const bf16_t* __restrict__ Vt, bf16_t* __restrict__ Ctx) {
    __shared__ __align__(32) bf16_t Plds[4 * 16 * 32];  // per-wave P scratch

    const int tid = threadIdx.x, w = tid >> 5, lane = tid & 31;
    const int mlo = lane & 15, half = lane >> 4;
    const int q0 = blockIdx.x * 64 + w * 16;
    const int h = blockIdx.y, b = blockIdx.z;
    const int kv = h >> 2;   // group size 4

    const bf16_t* Qb = Q  + ((size_t)(b * NH + h) * SEQ + q0) * HD;
    const bf16_t* Kp = Kb + (size_t)(b * NKV + kv) * SEQ * HD;
    const bf16_t* Vp = Vt + (size_t)(b * NKV + kv) * HD * SEQ;
    bf16_t* Pw = &Plds[w * 16 * 32];

    // Q fragments held in registers: 4 x (16x32) slabs of the 16x128 tile
    v16bf qf[4];
#pragma unroll
    for (int kq = 0; kq < 4; ++kq)
        qf[kq] = load_a_frag(Qb + mlo * HD + kq * 32, half);

    v8f o[8];
    float rmax[8], rsum[8];
#pragma unroll
    for (int dt = 0; dt < 8; ++dt)
        o[dt] = (v8f){0.f, 0.f, 0.f, 0.f, 0.f, 0.f, 0.f, 0.f};
#pragma unroll
    for (int v = 0; v < 8; ++v) { rmax[v] = -1.0e30f; rsum[v] = 0.0f; }

    const int nblk = (q0 + 16 + 31) >> 5;        // 32-key blocks up to diagonal
    for (int jb = 0; jb < nblk; ++jb) {
        const int key0 = jb * 32;
        v8f sc[2];
#pragma unroll
        for (int sub = 0; sub < 2; ++sub) {
            v8f c = (v8f){0.f, 0.f, 0.f, 0.f, 0.f, 0.f, 0.f, 0.f};
#pragma unroll
            for (int kq = 0; kq < 4; ++kq) {
                // B-frag of K^T: lane n = key-in-tile, elems = contiguous d
                const bf16_t* kp =
                    Kp + (size_t)(key0 + sub * 16 + mlo) * HD + kq * 32 + half * 16;
                c = wmma_bf16(qf[kq], *(const v16bf*)kp, c);
            }
            // causal mask: key index > query index  ->  -inf
#pragma unroll
            for (int v = 0; v < 8; ++v) {
                const int qi = q0 + v + 8 * half;
                const int ki = key0 + sub * 16 + mlo;
                if (ki > qi) c[v] = -3.0e38f;
            }
            sc[sub] = c;
        }
        // online softmax across this 32-key block
#pragma unroll
        for (int v = 0; v < 8; ++v) {
            float mx = fmaxf(sc[0][v], sc[1][v]);
#pragma unroll
            for (int msk = 8; msk >= 1; msk >>= 1) mx = fmaxf(mx, __shfl_xor(mx, msk));
            const float mnew = fmaxf(rmax[v], mx);
            const float corr = __expf(rmax[v] - mnew);
            const float p0 = __expf(sc[0][v] - mnew);
            const float p1 = __expf(sc[1][v] - mnew);
            float rs = p0 + p1;
#pragma unroll
            for (int msk = 8; msk >= 1; msk >>= 1) rs += __shfl_xor(rs, msk);
            rsum[v] = rsum[v] * corr + rs;
            rmax[v] = mnew;
#pragma unroll
            for (int dt = 0; dt < 8; ++dt) o[dt][v] *= corr;
            // scatter P (C-layout) into LDS as [m][k] for A-frag reload
            Pw[(v + 8 * half) * 32 + mlo]      = f2bf(p0);
            Pw[(v + 8 * half) * 32 + 16 + mlo] = f2bf(p1);
        }
        asm volatile("s_wait_dscnt 0x0" ::: "memory");  // same-wave LDS RAW fence

        const v16bf pf = load_a_frag(&Pw[mlo * 32], half);
#pragma unroll
        for (int dt = 0; dt < 8; ++dt) {
            // B-frag of V: lane n = d, elems = contiguous keys (V stored [d][s])
            const bf16_t* vp = Vp + (size_t)(dt * 16 + mlo) * SEQ + key0 + half * 16;
            o[dt] = wmma_bf16(pf, *(const v16bf*)vp, o[dt]);
        }
    }

    // normalize and write ctx [b][s][h*128 + d] as bf16
#pragma unroll
    for (int v = 0; v < 8; ++v) {
        const float inv = 1.0f / rsum[v];
        const int row = q0 + v + 8 * half;
        const size_t base = ((size_t)b * SEQ + row) * (size_t)(NH * HD) + h * HD;
#pragma unroll
        for (int dt = 0; dt < 8; ++dt)
            Ctx[base + dt * 16 + mlo] = f2bf(o[dt][v] * inv);
    }
}

// ---------------------------------------------------------------------------
// kernel_launch: x,mask,cos,sin,Wq,Wk,Wv,Wo,q_norm_w,k_norm_w -> out fp32
// ---------------------------------------------------------------------------
extern "C" void kernel_launch(void* const* d_in, const int* in_sizes, int n_in,
                              void* d_out, int out_size, void* d_ws, size_t ws_size,
                              hipStream_t stream) {
    const float* x    = (const float*)d_in[0];
    const float* cosb = (const float*)d_in[2];
    const float* sinb = (const float*)d_in[3];
    const float* Wq   = (const float*)d_in[4];
    const float* Wk   = (const float*)d_in[5];
    const float* Wv   = (const float*)d_in[6];
    const float* Wo   = (const float*)d_in[7];
    const float* qw   = (const float*)d_in[8];
    const float* kw   = (const float*)d_in[9];

    char* ws = (char*)d_ws;
    size_t off = 0;
    auto take = [&](size_t bytes) { char* p = ws + off; off += (bytes + 255) & ~(size_t)255; return p; };
    bf16_t* x_bf  = (bf16_t*)take((size_t)MROWS * 2048 * 2);
    bf16_t* wq_bf = (bf16_t*)take((size_t)2048 * 2048 * 2);
    bf16_t* wk_bf = (bf16_t*)take((size_t)2048 * 512 * 2);
    bf16_t* wv_bf = (bf16_t*)take((size_t)2048 * 512 * 2);
    bf16_t* wo_bf = (bf16_t*)take((size_t)2048 * 2048 * 2);
    float*  q_raw = (float*)take((size_t)MROWS * 2048 * 4);
    float*  k_raw = (float*)take((size_t)MROWS * 512 * 4);
    float*  v_raw = (float*)take((size_t)MROWS * 512 * 4);
    bf16_t* q_bf  = (bf16_t*)take((size_t)MROWS * 2048 * 2);
    bf16_t* k_bf  = (bf16_t*)take((size_t)MROWS * 512 * 2);
    bf16_t* v_t   = (bf16_t*)take((size_t)MROWS * 512 * 2);
    bf16_t* ctx   = (bf16_t*)take((size_t)MROWS * 2048 * 2);

    // 0) one-shot fp32 -> bf16 conversions
    auto cvt = [&](const float* src, bf16_t* dst, size_t n) {
        int n8 = (int)(n / 8);
        f32_to_bf16<<<(n8 + 255) / 256, 256, 0, stream>>>(src, dst, n8);
    };
    cvt(x,  x_bf,  (size_t)MROWS * 2048);
    cvt(Wq, wq_bf, (size_t)2048 * 2048);
    cvt(Wk, wk_bf, (size_t)2048 * 512);
    cvt(Wv, wv_bf, (size_t)2048 * 512);
    cvt(Wo, wo_bf, (size_t)2048 * 2048);

    // 1) QKV projections (bf16 x bf16 -> fp32 raw outputs)
    gemm_wmma<2048, EMB><<<dim3(2048 / 128, MROWS / 128), 256, 0, stream>>>(x_bf, wq_bf, q_raw);
    gemm_wmma<512,  EMB><<<dim3(512 / 128,  MROWS / 128), 256, 0, stream>>>(x_bf, wk_bf, k_raw);
    gemm_wmma<512,  EMB><<<dim3(512 / 128,  MROWS / 128), 256, 0, stream>>>(x_bf, wv_bf, v_raw);

    // 2) RMSNorm + RoPE, repack to bf16 (v transposed for P.V fragments)
    norm_rope<<<dim3(MROWS, NH + 2 * NKV), 128, 0, stream>>>(
        q_raw, k_raw, v_raw, cosb, sinb, qw, kw, q_bf, k_bf, v_t);

    // 3) causal flash attention -> ctx bf16
    attn_wmma<<<dim3(SEQ / 64, NH, BATCH), 128, 0, stream>>>(q_bf, k_bf, v_t, ctx);

    // 4) output projection (bf16 ctx x bf16 Wo -> fp32 out)
    gemm_wmma<2048, 2048><<<dim3(2048 / 128, MROWS / 128), 256, 0, stream>>>(ctx, wo_bf, (float*)d_out);
}
